// GCN_481036337794
// MI455X (gfx1250) — matrix operand order repaired and checked by reading, not compiled, not run on hardware
//
#include <hip/hip_runtime.h>
#include <hip/hip_bf16.h>
#include <cstdint>

typedef __attribute__((ext_vector_type(16))) _Float16 v16h;
typedef __attribute__((ext_vector_type(8)))  _Float16 v8h;
typedef __attribute__((ext_vector_type(8)))  float    v8f;

#define N_NODES_C 16384
#define N_GRAPHS_C 256
#define EDGES_PER_GRAPH_C 2048
#define N_EDGES_C (N_GRAPHS_C * EDGES_PER_GRAPH_C)
#define SIG_LEN_C 100
#define CCH 64          // encoder channels
#define L2LEN 20        // length after first maxpool
#define NCOLS (N_NODES_C * L2LEN)   // 327680 (node,pos) columns
#define DH 128          // processor hidden dim

// ---------------------------------------------------------------------------
// WMMA fragment loaders (layouts per CDNA5 ISA 7.12.2, wave32).
// A tile in LDS as [m][k] (ldk halves): lane l -> row m0+(l&15),
//   k runs {k0+(l>>4)*8 .. +7} and {+16 .. +23}: two contiguous v8h loads.
// B tile in LDS TRANSPOSED as [n][k]: lane l -> col n0+(l&15),
//   k runs {k0+(l>>4)*8 .. +7} and {+16 .. +23}: two contiguous v8h loads.
// C/D: v8f, lane l, reg r: M = r + (l>>4)*8, N = l&15.
// ---------------------------------------------------------------------------
__device__ __forceinline__ v16h concat8(v8h lo, v8h hi) {
  return __builtin_shufflevector(lo, hi, 0, 1, 2, 3, 4, 5, 6, 7, 8, 9, 10, 11,
                                 12, 13, 14, 15);
}

__device__ __forceinline__ v16h load_A_frag(const _Float16* As, int m0, int k0,
                                            int ldk, int lane) {
  const int m = m0 + (lane & 15);
  const int kb = k0 + ((lane >> 4) << 3);
  const v8h r0 = *(const v8h*)(As + m * ldk + kb);
  const v8h r1 = *(const v8h*)(As + m * ldk + kb + 16);
  return concat8(r0, r1);
}

__device__ __forceinline__ v16h load_B_frag(const _Float16* BsT, int n0, int k0,
                                            int ldk, int lane) {
  const int n = n0 + (lane & 15);
  const int kb = k0 + ((lane >> 4) << 3);
  const v8h r0 = *(const v8h*)(BsT + n * ldk + kb);
  const v8h r1 = *(const v8h*)(BsT + n * ldk + kb + 16);
  return concat8(r0, r1);
}

__device__ __forceinline__ v8f wmma_f16(v16h a, v16h b, v8f c) {
  return __builtin_amdgcn_wmma_f32_16x16x32_f16(false, a, false, b, (short)0, c,
                                                false, false);
}

// ---------------------------------------------------------------------------
// Encoder stage 0: conv1d(1->64, k7, pad3). Two-pass BN (stats then apply).
// ---------------------------------------------------------------------------
__global__ void conv0_stats_kernel(const float* __restrict__ x,
                                   const float* __restrict__ w0,
                                   const float* __restrict__ b0,
                                   float* __restrict__ sums,
                                   float* __restrict__ sumsq) {
  __shared__ float xs[SIG_LEN_C];
  const int node = blockIdx.x;
  const int c = threadIdx.x;  // 64 threads = 64 channels
  for (int i = threadIdx.x; i < SIG_LEN_C; i += 64) xs[i] = x[node * SIG_LEN_C + i];
  __syncthreads();
  float wk[7];
#pragma unroll
  for (int t = 0; t < 7; ++t) wk[t] = w0[c * 7 + t];
  const float bb = b0[c];
  float s = 0.f, s2 = 0.f;
  for (int p = 0; p < SIG_LEN_C; ++p) {
    float acc = bb;
#pragma unroll
    for (int t = 0; t < 7; ++t) {
      const int ip = p + t - 3;
      if (ip >= 0 && ip < SIG_LEN_C) acc += wk[t] * xs[ip];
    }
    s += acc;
    s2 += acc * acc;
  }
  atomicAdd(&sums[c], s);
  atomicAdd(&sumsq[c], s2);
}

__global__ void finalize_stats_kernel(const float* __restrict__ sums,
                                      const float* __restrict__ sumsq,
                                      const float* __restrict__ g,
                                      const float* __restrict__ be,
                                      float* __restrict__ scale,
                                      float* __restrict__ shift, int C,
                                      float invCount) {
  const int c = blockIdx.x * blockDim.x + threadIdx.x;
  if (c >= C) return;
  const float m = sums[c] * invCount;
  const float v = sumsq[c] * invCount - m * m;
  const float sc = g[c] * rsqrtf(v + 1e-5f);
  scale[c] = sc;
  shift[c] = be[c] - m * sc;
}

// recompute conv0, apply BN+ReLU, maxpool5 -> z[(node*20+pp)*64 + c] (f16)
__global__ void conv0_apply_kernel(const float* __restrict__ x,
                                   const float* __restrict__ w0,
                                   const float* __restrict__ b0,
                                   const float* __restrict__ scale,
                                   const float* __restrict__ shift,
                                   _Float16* __restrict__ z) {
  __shared__ float xs[SIG_LEN_C];
  const int node = blockIdx.x;
  const int c = threadIdx.x;
  for (int i = threadIdx.x; i < SIG_LEN_C; i += 64) xs[i] = x[node * SIG_LEN_C + i];
  __syncthreads();
  float wk[7];
#pragma unroll
  for (int t = 0; t < 7; ++t) wk[t] = w0[c * 7 + t];
  const float bb = b0[c];
  const float sc = scale[c], sh = shift[c];
  for (int pp = 0; pp < L2LEN; ++pp) {
    float mx = 0.f;  // post-ReLU values are >= 0
#pragma unroll
    for (int q = 0; q < 5; ++q) {
      const int p = pp * 5 + q;
      float acc = bb;
#pragma unroll
      for (int t = 0; t < 7; ++t) {
        const int ip = p + t - 3;
        if (ip >= 0 && ip < SIG_LEN_C) acc += wk[t] * xs[ip];
      }
      mx = fmaxf(mx, fmaxf(acc * sc + sh, 0.f));
    }
    z[((size_t)node * L2LEN + pp) * CCH + c] = (_Float16)mx;
  }
}

// convert conv weights (64x64x5 f32, tap innermost) -> wtap[tap][m][kin] f16
__global__ void prep_wconv_kernel(const float* __restrict__ w,
                                  _Float16* __restrict__ wtap) {
  const int i = blockIdx.x * 256 + threadIdx.x;  // 5*64*64 = 20480
  if (i >= 5 * CCH * CCH) return;
  const int tap = i / (CCH * CCH);
  const int r = i % (CCH * CCH);
  const int m = r >> 6, kin = r & 63;
  wtap[i] = (_Float16)w[(m * CCH + kin) * 5 + tap];
}

// ---------------------------------------------------------------------------
// conv1/conv2 as WMMA GEMM: y[(node,pos), outch] = W(64x320) * patches(320 x cols)
// K chunked by tap (5 chunks of 64). Block tile: 64 outch x 64 cols, 8 waves.
// ---------------------------------------------------------------------------
__global__ void conv_gemm_kernel(const _Float16* __restrict__ zin,
                                 const _Float16* __restrict__ wtap,  // [5][64][64]
                                 const float* __restrict__ bias,
                                 float* __restrict__ y) {
  __shared__ __align__(16) _Float16 As[CCH * 64];   // [m][k-chunk]
  __shared__ __align__(16) _Float16 BsT[64 * 64];   // [col][k-chunk] (transposed)
  const int lane = threadIdx.x & 31;
  const int wave = threadIdx.x >> 5;
  const int col0 = blockIdx.x * 64;
  v8f acc0 = {};
  v8f acc1 = {};
  const int m0 = (wave & 3) * 16;
  const int nb = (wave >> 2) * 32;
  for (int tap = 0; tap < 5; ++tap) {
    // A chunk: 512 v8h copies (contiguous f16 source)
    for (int i = threadIdx.x; i < (CCH * 64) / 8; i += 256) {
      *(v8h*)&As[i * 8] = *(const v8h*)&wtap[tap * CCH * CCH + i * 8];
    }
    // B chunk (transposed): per column one contiguous 64-half row of zin
    for (int i = threadIdx.x; i < (64 * 64) / 8; i += 256) {
      const int col = i >> 3;
      const int k8 = (i & 7) * 8;
      const int c = col0 + col;
      const int node = c / L2LEN, pos = c % L2LEN;
      const int sp = pos + tap - 2;
      v8h val = {};
      if (sp >= 0 && sp < L2LEN)
        val = *(const v8h*)&zin[((size_t)node * L2LEN + sp) * CCH + k8];
      *(v8h*)&BsT[col * 64 + k8] = val;
    }
    if (tap < 4) {  // prefetch next tap's activations (global_prefetch_b8)
      const int c = col0 + (threadIdx.x & 63);
      const int node = c / L2LEN, pos = c % L2LEN;
      const int sp = pos + tap - 1;
      if (sp >= 0 && sp < L2LEN)
        __builtin_prefetch(&zin[((size_t)node * L2LEN + sp) * CCH], 0, 1);
    }
    __syncthreads();
#pragma unroll
    for (int kk = 0; kk < 64; kk += 32) {
      const v16h a = load_A_frag(As, m0, kk, 64, lane);
      const v16h b0 = load_B_frag(BsT, nb, kk, 64, lane);
      const v16h b1 = load_B_frag(BsT, nb + 16, kk, 64, lane);
      acc0 = wmma_f16(a, b0, acc0);
      acc1 = wmma_f16(a, b1, acc1);
    }
    __syncthreads();
  }
  const int mrow = m0 + ((lane >> 4) << 3);
  const int n = lane & 15;
#pragma unroll
  for (int r = 0; r < 8; ++r) {
    const int m = mrow + r;
    const float bv = bias[m];
    y[(size_t)(col0 + nb + n) * CCH + m] = acc0[r] + bv;
    y[(size_t)(col0 + nb + 16 + n) * CCH + m] = acc1[r] + bv;
  }
}

// per-channel sum/sumsq over rows of a row-major [rows][C] f32 matrix
__global__ void col_stats_kernel(const float* __restrict__ y, int nrows, int C,
                                 float* __restrict__ sums,
                                 float* __restrict__ sumsq) {
  const int c = threadIdx.x;  // blockDim.x == C
  float s = 0.f, s2 = 0.f;
  for (int r = blockIdx.x; r < nrows; r += gridDim.x) {
    const float v = y[(size_t)r * C + c];
    s += v;
    s2 += v * v;
  }
  atomicAdd(&sums[c], s);
  atomicAdd(&sumsq[c], s2);
}

// z_out = relu(relu(scale*y+shift) + z_in)   (residual conv layers)
__global__ void bn_relu_res_kernel(const float* __restrict__ y,
                                   const _Float16* __restrict__ zin,
                                   _Float16* __restrict__ zout,
                                   const float* __restrict__ scale,
                                   const float* __restrict__ shift, int n) {
  const int i = blockIdx.x * blockDim.x + threadIdx.x;
  if (i >= n) return;
  const int c = i & (CCH - 1);
  const float h = fmaxf(y[(size_t)i] * scale[c] + shift[c], 0.f);
  zout[(size_t)i] = (_Float16)fmaxf(h + (float)zin[(size_t)i], 0.f);
}

// ---------------------------------------------------------------------------
// conv3 (64->8, k5) VALU (tiny), then BN+ReLU+maxpool5 -> 32-dim embedding
// ---------------------------------------------------------------------------
__global__ void conv3_kernel(const _Float16* __restrict__ zin,
                             const float* __restrict__ w3,  // [8][64][5]
                             const float* __restrict__ b3,
                             float* __restrict__ y3) {
  __shared__ _Float16 zs[L2LEN * CCH];
  const int node = blockIdx.x;
  for (int i = threadIdx.x; i < L2LEN * CCH; i += 64)
    zs[i] = zin[(size_t)node * L2LEN * CCH + i];
  __syncthreads();
  const int oc = threadIdx.x & 7;
  const int pbase = threadIdx.x >> 3;  // 8 position groups
  for (int p = pbase; p < L2LEN; p += 8) {
    float acc = b3[oc];
#pragma unroll
    for (int t = 0; t < 5; ++t) {
      const int sp = p + t - 2;
      if (sp < 0 || sp >= L2LEN) continue;
      for (int ic = 0; ic < CCH; ++ic)
        acc += w3[(oc * CCH + ic) * 5 + t] * (float)zs[sp * CCH + ic];
    }
    y3[((size_t)node * L2LEN + p) * 8 + oc] = acc;
  }
}

__global__ void conv3_pool_kernel(const float* __restrict__ y3,
                                  const float* __restrict__ scale,
                                  const float* __restrict__ shift,
                                  float* __restrict__ h0) {
  const int node = blockIdx.x;
  const int t = threadIdx.x;  // 32 threads: feature = ch*4 + pp
  const int ch = t >> 2, pp = t & 3;
  float mx = 0.f;
#pragma unroll
  for (int q = 0; q < 5; ++q) {
    const int p = pp * 5 + q;
    const float v =
        fmaxf(y3[((size_t)node * L2LEN + p) * 8 + ch] * scale[ch] + shift[ch], 0.f);
    mx = fmaxf(mx, v);
  }
  h0[node * 32 + t] = mx;
}

// ---------------------------------------------------------------------------
// Processor: per-graph scatter-add in LDS, then fused [agg||x] @ [Wr;Wo]^T GEMM
// ---------------------------------------------------------------------------
__global__ void prep_wcat_kernel(const float* __restrict__ Wr,
                                 const float* __restrict__ Wo,
                                 _Float16* __restrict__ wcat, int din) {
  const int i = blockIdx.x * blockDim.x + threadIdx.x;
  if (i >= DH * din) return;
  const int m = i / din, k = i % din;
  wcat[m * (2 * din) + k] = (_Float16)Wr[i];
  wcat[m * (2 * din) + din + k] = (_Float16)Wo[i];
}

__global__ void aggregate_kernel(const float* __restrict__ h,
                                 const int* __restrict__ src,
                                 const int* __restrict__ dst,
                                 const float* __restrict__ ew,
                                 _Float16* __restrict__ feat, int din) {
  extern __shared__ float smem[];
  float* xg = smem;             // [64][din]
  float* ag = smem + 64 * din;  // [64][din]
  const int g = blockIdx.x;
  for (int i = threadIdx.x; i < 64 * din; i += 256) {
    xg[i] = h[(size_t)g * 64 * din + i];
    ag[i] = 0.f;
  }
  __syncthreads();
  const int lane = threadIdx.x & 31;
  const int wave = threadIdx.x >> 5;
  const int e0 = g * EDGES_PER_GRAPH_C + wave * (EDGES_PER_GRAPH_C / 8);
  for (int e = e0; e < e0 + EDGES_PER_GRAPH_C / 8; ++e) {
    const int s = src[e] - g * 64;
    const int d = dst[e] - g * 64;
    const float w = ew[e];
    for (int f = lane; f < din; f += 32)
      atomicAdd(&ag[d * din + f], xg[s * din + f] * w);
  }
  __syncthreads();
  for (int i = threadIdx.x; i < 64 * din; i += 256) {
    const int n = i / din, f = i % din;
    const size_t row = (size_t)(g * 64 + n) * (2 * din);
    feat[row + f] = (_Float16)ag[i];
    feat[row + din + f] = (_Float16)xg[i];
  }
}

// y[node][m] = sum_k feat[node][k] * wcat[m][k] + br[m]; block = 64 nodes, 8 waves
__global__ void node_gemm_kernel(const _Float16* __restrict__ feat,
                                 const _Float16* __restrict__ wcat,
                                 const float* __restrict__ br,
                                 float* __restrict__ y, int K) {
  __shared__ __align__(16) _Float16 As[DH * 64];   // [m][k-chunk]
  __shared__ __align__(16) _Float16 BsT[64 * 64];  // [node][k-chunk] (transposed)
  const int lane = threadIdx.x & 31;
  const int wave = threadIdx.x >> 5;
  const int node0 = blockIdx.x * 64;
  const int m0 = wave * 16;
  v8f acc[4] = {};
  for (int kc = 0; kc < K; kc += 64) {
    for (int i = threadIdx.x; i < (DH * 64) / 8; i += 256) {
      const int m = i >> 3;
      const int k8 = (i & 7) * 8;
      *(v8h*)&As[m * 64 + k8] = *(const v8h*)&wcat[m * K + kc + k8];
    }
    for (int i = threadIdx.x; i < (64 * 64) / 8; i += 256) {
      const int n = i >> 3;
      const int k8 = (i & 7) * 8;
      *(v8h*)&BsT[n * 64 + k8] = *(const v8h*)&feat[(size_t)(node0 + n) * K + kc + k8];
    }
    if (kc + 64 < K)  // prefetch next chunk of node features
      __builtin_prefetch(&feat[(size_t)(node0 + (threadIdx.x & 63)) * K + kc + 64], 0, 1);
    __syncthreads();
#pragma unroll
    for (int kk = 0; kk < 64; kk += 32) {
      const v16h a = load_A_frag(As, m0, kk, 64, lane);
#pragma unroll
      for (int nt = 0; nt < 4; ++nt) {
        const v16h b = load_B_frag(BsT, nt * 16, kk, 64, lane);
        acc[nt] = wmma_f16(a, b, acc[nt]);
      }
    }
    __syncthreads();
  }
  const int mrow = m0 + ((lane >> 4) << 3);
  const int n = lane & 15;
#pragma unroll
  for (int nt = 0; nt < 4; ++nt) {
    const int nn = node0 + nt * 16 + n;
#pragma unroll
    for (int r = 0; r < 8; ++r)
      y[(size_t)nn * DH + mrow + r] = acc[nt][r] + br[mrow + r];
  }
}

__global__ void node_bn_kernel(const float* __restrict__ y,
                               const float* __restrict__ hprev,
                               float* __restrict__ hout,
                               const float* __restrict__ scale,
                               const float* __restrict__ shift, int n,
                               int residual) {
  const int i = blockIdx.x * blockDim.x + threadIdx.x;
  if (i >= n) return;
  const int c = i & (DH - 1);
  float o = fmaxf(y[(size_t)i] * scale[c] + shift[c], 0.f);
  if (residual) o = fmaxf(o + hprev[(size_t)i], 0.f);
  hout[(size_t)i] = o;
}

// ---------------------------------------------------------------------------
// Per-graph mean pool + MLP head (128 -> 64 -> LN -> ReLU -> 3 -> sigmoid)
// ---------------------------------------------------------------------------
__global__ void classifier_kernel(const float* __restrict__ h,
                                  const float* __restrict__ W1,
                                  const float* __restrict__ b1,
                                  const float* __restrict__ ln_g,
                                  const float* __restrict__ ln_b,
                                  const float* __restrict__ W2,
                                  const float* __restrict__ b2,
                                  float* __restrict__ out) {
  __shared__ float hg[DH];
  __shared__ float u[64];
  __shared__ float red[2];
  const int g = blockIdx.x;
  const int t = threadIdx.x;  // 128 threads
  float s = 0.f;
  for (int nl = 0; nl < 64; ++nl) s += h[((size_t)g * 64 + nl) * DH + t];
  hg[t] = s * (1.f / 64.f);
  __syncthreads();
  if (t < 64) {
    float acc = b1[t];
    for (int k = 0; k < DH; ++k) acc += W1[t * DH + k] * hg[k];
    u[t] = acc;
  }
  __syncthreads();
  if (t == 0) {
    float m = 0.f;
    for (int j = 0; j < 64; ++j) m += u[j];
    m *= (1.f / 64.f);
    float v = 0.f;
    for (int j = 0; j < 64; ++j) { const float d = u[j] - m; v += d * d; }
    v *= (1.f / 64.f);
    red[0] = m;
    red[1] = rsqrtf(v + 1e-5f);
  }
  __syncthreads();
  if (t < 64) u[t] = fmaxf((u[t] - red[0]) * red[1] * ln_g[t] + ln_b[t], 0.f);
  __syncthreads();
  if (t < 3) {
    float acc = b2[t];
    for (int k = 0; k < 64; ++k) acc += W2[t * 64 + k] * u[k];
    out[g * 3 + t] = 1.f / (1.f + expf(-acc));
  }
}

// ---------------------------------------------------------------------------
extern "C" void kernel_launch(void* const* d_in, const int* in_sizes, int n_in,
                              void* d_out, int out_size, void* d_ws, size_t ws_size,
                              hipStream_t stream) {
  (void)in_sizes; (void)n_in; (void)out_size; (void)ws_size;
  const float* x  = (const float*)d_in[0];
  const int* ei   = (const int*)d_in[1];
  const float* ew = (const float*)d_in[2];
  // d_in[3] = batch (implicit: node/64)
  const float* w0 = (const float*)d_in[4];  const float* b0 = (const float*)d_in[5];
  const float* g0 = (const float*)d_in[6];  const float* be0 = (const float*)d_in[7];
  const float* wc[2] = {(const float*)d_in[8], (const float*)d_in[12]};
  const float* bc[2] = {(const float*)d_in[9], (const float*)d_in[13]};
  const float* gc[2] = {(const float*)d_in[10], (const float*)d_in[14]};
  const float* bec[2] = {(const float*)d_in[11], (const float*)d_in[15]};
  const float* w3 = (const float*)d_in[16]; const float* b3 = (const float*)d_in[17];
  const float* g3 = (const float*)d_in[18]; const float* be3 = (const float*)d_in[19];
  const float* W1 = (const float*)d_in[40]; const float* cb1 = (const float*)d_in[41];
  const float* lng = (const float*)d_in[42]; const float* lnb = (const float*)d_in[43];
  const float* W2 = (const float*)d_in[44]; const float* cb2 = (const float*)d_in[45];
  float* out = (float*)d_out;

  // workspace carve-up
  char* ws = (char*)d_ws;
  const size_t ZBYTES = (size_t)NCOLS * CCH * sizeof(_Float16);  // 41.9 MB
  _Float16* z_a = (_Float16*)ws;                 ws += ZBYTES;
  _Float16* z_b = (_Float16*)ws;                 ws += ZBYTES;
  float* y      = (float*)ws;                    ws += (size_t)NCOLS * CCH * sizeof(float);
  _Float16* feat = (_Float16*)ws;                ws += (size_t)N_NODES_C * 256 * sizeof(_Float16);
  float* h_a    = (float*)ws;                    ws += (size_t)N_NODES_C * DH * sizeof(float);
  float* h_b    = (float*)ws;                    ws += (size_t)N_NODES_C * DH * sizeof(float);
  _Float16* wbuf = (_Float16*)ws;                ws += (size_t)DH * 256 * sizeof(_Float16); // wcat / wtap
  float* sums   = (float*)ws;                    ws += 128 * sizeof(float);
  float* sumsq  = (float*)ws;                    ws += 128 * sizeof(float);
  float* scale  = (float*)ws;                    ws += 128 * sizeof(float);
  float* shift  = (float*)ws;                    ws += 128 * sizeof(float);

  const int* src = ei;
  const int* dst = ei + N_EDGES_C;

  // ---- encoder stage 0 ----
  hipMemsetAsync(sums, 0, 2 * 128 * sizeof(float), stream);
  conv0_stats_kernel<<<N_NODES_C, 64, 0, stream>>>(x, w0, b0, sums, sumsq);
  finalize_stats_kernel<<<1, 64, 0, stream>>>(sums, sumsq, g0, be0, scale, shift,
                                              CCH, 1.f / (float)(N_NODES_C * SIG_LEN_C));
  conv0_apply_kernel<<<N_NODES_C, 64, 0, stream>>>(x, w0, b0, scale, shift, z_a);

  // ---- residual conv layers 1 & 2 (WMMA GEMM) ----
  _Float16* zc = z_a;
  _Float16* zn = z_b;
  for (int i = 0; i < 2; ++i) {
    prep_wconv_kernel<<<80, 256, 0, stream>>>(wc[i], wbuf);
    conv_gemm_kernel<<<NCOLS / 64, 256, 0, stream>>>(zc, wbuf, bc[i], y);
    hipMemsetAsync(sums, 0, 2 * 128 * sizeof(float), stream);
    col_stats_kernel<<<256, CCH, 0, stream>>>(y, NCOLS, CCH, sums, sumsq);
    finalize_stats_kernel<<<1, 64, 0, stream>>>(sums, sumsq, gc[i], bec[i], scale,
                                                shift, CCH, 1.f / (float)NCOLS);
    bn_relu_res_kernel<<<(NCOLS * CCH) / 256, 256, 0, stream>>>(y, zc, zn, scale,
                                                                shift, NCOLS * CCH);
    _Float16* tmp = zc; zc = zn; zn = tmp;
  }

  // ---- conv3 + pool -> 32-dim node embedding in h_a ----
  conv3_kernel<<<N_NODES_C, 64, 0, stream>>>(zc, w3, b3, y);
  hipMemsetAsync(sums, 0, 2 * 128 * sizeof(float), stream);
  col_stats_kernel<<<256, 8, 0, stream>>>(y, NCOLS, 8, sums, sumsq);
  finalize_stats_kernel<<<1, 8, 0, stream>>>(sums, sumsq, g3, be3, scale, shift, 8,
                                             1.f / (float)NCOLS);
  conv3_pool_kernel<<<N_NODES_C, 32, 0, stream>>>(y, scale, shift, h_a);

  // ---- processor: 4 GCN layers ----
  float* hc = h_a;
  float* hn = h_b;
  for (int i = 0; i < 4; ++i) {
    const int din = (i == 0) ? 32 : DH;
    const int K = 2 * din;
    const float* Wr = (const float*)d_in[20 + 5 * i];
    const float* br = (const float*)d_in[21 + 5 * i];
    const float* Wo = (const float*)d_in[22 + 5 * i];
    const float* gn = (const float*)d_in[23 + 5 * i];
    const float* bn = (const float*)d_in[24 + 5 * i];
    prep_wcat_kernel<<<(DH * din + 255) / 256, 256, 0, stream>>>(Wr, Wo, wbuf, din);
    aggregate_kernel<<<N_GRAPHS_C, 256, (size_t)(2 * 64 * din) * sizeof(float), stream>>>(
        hc, src, dst, ew, feat, din);
    node_gemm_kernel<<<N_NODES_C / 64, 256, 0, stream>>>(feat, wbuf, br, y, K);
    hipMemsetAsync(sums, 0, 2 * 128 * sizeof(float), stream);
    col_stats_kernel<<<256, DH, 0, stream>>>(y, N_NODES_C, DH, sums, sumsq);
    finalize_stats_kernel<<<1, 128, 0, stream>>>(sums, sumsq, gn, bn, scale, shift,
                                                 DH, 1.f / (float)N_NODES_C);
    node_bn_kernel<<<(N_NODES_C * DH) / 256, 256, 0, stream>>>(
        y, hc, hn, scale, shift, N_NODES_C * DH, (i == 1 || i == 2) ? 1 : 0);
    float* tmp = hc; hc = hn; hn = tmp;
  }

  // ---- classifier head ----
  classifier_kernel<<<N_GRAPHS_C, 128, 0, stream>>>(hc, W1, cb1, lng, lnb, W2, cb2, out);
}
